// KNNKPFNodeMatcher_35218731827997
// MI455X (gfx1250) — matrix operand order repaired and checked by reading, not compiled, not run on hardware
//
#include <hip/hip_runtime.h>
#include <hip/hip_bf16.h>

// ---------------------------------------------------------------------------
// KNN/KPF RANSAC node matcher for MI455X (gfx1250), wave32 + WMMA f16.
//
// Pipeline (all on `stream`, deterministic):
//  1. k_pre1    : f32->f16 casts of x,y; y transposed copy; ||y_m||^2
//  2. k_mask    : w = qmask & kmask (f16 0/1), row counts rc[n]
//  3. k_cols    : column counts cc[m]
//  4. k_sums    : cnt[s] = sum rc, bsum[s] = sum cc*||y||^2
//  5. k_gather  : gathered transposed sample matrices sx^T (per it), sy^T (per s,it)
//  6. k_U       : U = W @ y per s          (WMMA)
//  7. k_solver  : model = sx^T W_s sy /den (WMMA, 2-stage GEMM, LDS-staged W chunk)
//  8. k_avg     : xt = x@model, closed-form masked mean error (WMMA)
//  9. k_best    : argmin over IT
// 10. k_xtn2    : ||xt_n||^2 for winning models
// 11. k_err     : full error grid for winners: xtn2+ynorm-2*xt@y^T, NaN off-mask (WMMA)
// 12. k_quant   : exact nanquantile(0.75) via 4-pass radix select on float keys
// 13. k_order   : top_k (=S) descending sort of per-slice means -> output errs
// 14. k_tm/k_ti : reorder models + thresholded inlier masks into d_out
// ---------------------------------------------------------------------------

typedef _Float16 v16h __attribute__((ext_vector_type(16)));
typedef _Float16 v8h  __attribute__((ext_vector_type(8)));
typedef float    v8f  __attribute__((ext_vector_type(8)));

#define SD   8
#define ITD  16
#define ND   768
#define MD   768
#define DD   64
#define RN   460
#define RP   480          // padded sample count (zero fill)
#define NMD  (ND*MD)      // 589824
#define NCHUNK 15         // 480/32 i-chunks in solver
#define KSA  15           // 480/32 k-steps in solver step A

__device__ __forceinline__ float nanf_dev() { return __uint_as_float(0x7FC00000u); }

__device__ __forceinline__ v8f vzero8() {
  v8f z;
#pragma unroll
  for (int r = 0; r < 8; ++r) z[r] = 0.0f;
  return z;
}

// A fragment (16x32 f16, M x K) from row-major [row][K] storage.
// Layout: lane L holds row M=L&15; element h maps to K = h + 8*(h>=8) + 8*(L>=16).
__device__ __forceinline__ v16h load_a_frag(const _Float16* base, int stride,
                                            int row, int kbase, int hi) {
  const _Float16* p = base + row * stride + kbase + hi * 8;
  v8h lo = *(const v8h*)(p);
  v8h hh = *(const v8h*)(p + 16);
  v16h a;
#pragma unroll
  for (int h = 0; h < 8; ++h) { a[h] = lo[h]; a[h + 8] = hh[h]; }
  return a;
}

// B fragment (32x16 f16, K x N) from K-contiguous [col][K] (i.e. B^T) storage.
// Layout: lane L holds col N=L&15; element h maps to K = h + 8*(h>=8) + 8*(L>=16).
__device__ __forceinline__ v16h load_b_frag(const _Float16* base, int stride,
                                            int col, int kbase, int hi) {
  const _Float16* p = base + col * stride + kbase + hi * 8;
  v8h lo = *(const v8h*)(p);
  v8h hh = *(const v8h*)(p + 16);
  v16h b;
#pragma unroll
  for (int h = 0; h < 8; ++h) { b[h] = lo[h]; b[h + 8] = hh[h]; }
  return b;
}

__device__ __forceinline__ v8f wmma16(v16h a, v16h b, v8f c) {
  return __builtin_amdgcn_wmma_f32_16x16x32_f16(false, a, false, b, (short)0, c,
                                                false, false);
}

__device__ __forceinline__ float block_reduce_256(float v, float* red) {
  int tid = threadIdx.x;
  red[tid] = v;
  __syncthreads();
#pragma unroll
  for (int s = 128; s > 0; s >>= 1) {
    if (tid < s) red[tid] += red[tid + s];
    __syncthreads();
  }
  float r = red[0];
  __syncthreads();
  return r;
}

// ---------------------------------------------------------------------------
__global__ void k_pre1(const float* __restrict__ qk, const float* __restrict__ kks,
                       _Float16* __restrict__ x_h, _Float16* __restrict__ y_h,
                       _Float16* __restrict__ yT_h, float* __restrict__ ynorm) {
  int t = blockIdx.x * 256 + threadIdx.x;
  if (t < SD * MD) {
    int s = t / MD, m = t % MD;
    float acc = 0.0f;
#pragma unroll 4
    for (int e = 0; e < DD; ++e) {
      float v = kks[(s * MD + m) * DD + e];
      y_h[(s * MD + m) * DD + e] = (_Float16)v;
      yT_h[(s * DD + e) * MD + m] = (_Float16)v;
      acc += v * v;
    }
    ynorm[t] = acc;
  } else if (t < SD * MD + ND) {
    int n = t - SD * MD;
#pragma unroll 4
    for (int d = 0; d < DD; ++d) x_h[n * DD + d] = (_Float16)qk[n * DD + d];
  }
}

__global__ void k_mask(const int* __restrict__ qm, const int* __restrict__ km,
                       _Float16* __restrict__ w_h, float* __restrict__ rc) {
  int t = blockIdx.x * 256 + threadIdx.x;
  if (t >= SD * ND) return;
  long base = (long)t * MD;
  float acc = 0.0f;
  for (int m = 0; m < MD; ++m) {
    int w = (qm[base + m] != 0) && (km[base + m] != 0);
    w_h[base + m] = (_Float16)(float)w;
    acc += (float)w;
  }
  rc[t] = acc;
}

__global__ void k_cols(const _Float16* __restrict__ w_h, float* __restrict__ cc) {
  int t = blockIdx.x * 256 + threadIdx.x;
  if (t >= SD * MD) return;
  int s = t / MD, m = t % MD;
  float acc = 0.0f;
  for (int n = 0; n < ND; ++n) acc += (float)w_h[((long)s * ND + n) * MD + m];
  cc[t] = acc;
}

__global__ void k_sums(const float* __restrict__ rc, const float* __restrict__ cc,
                       const float* __restrict__ ynorm, float* __restrict__ cnt,
                       float* __restrict__ bsum) {
  __shared__ float red[256];
  int s = blockIdx.x, tid = threadIdx.x;
  float pc = 0.0f, pb = 0.0f;
  for (int i = tid; i < ND; i += 256) {
    pc += rc[s * ND + i];
    pb += cc[s * MD + i] * ynorm[s * MD + i];
  }
  float c = block_reduce_256(pc, red);
  float b = block_reduce_256(pb, red);
  if (tid == 0) { cnt[s] = c; bsum[s] = b; }
}

__global__ void k_gather(const int* __restrict__ perm1, const int* __restrict__ perm2,
                         const _Float16* __restrict__ x_h, const _Float16* __restrict__ y_h,
                         _Float16* __restrict__ sxT, _Float16* __restrict__ syT) {
  int t = blockIdx.x * 256 + threadIdx.x;
  const int T1 = ITD * DD * RP;
  const int T2 = SD * ITD * DD * RP;
  if (t < T1) {
    int it = t / (DD * RP);
    int r = t % (DD * RP);
    int d = r / RP, i = r % RP;
    sxT[t] = (i < RN) ? x_h[perm1[it * RN + i] * DD + d] : (_Float16)0.0f;
  } else if (t < T1 + T2) {
    int u = t - T1;
    int blk = u / (DD * RP);
    int r = u % (DD * RP);
    int e = r / RP, j = r % RP;
    int s = blk / ITD, it = blk % ITD;
    syT[u] = (j < RN) ? y_h[((long)s * MD + perm2[it * RN + j]) * DD + e]
                      : (_Float16)0.0f;
  }
}

// U[s][n][e] = sum_m w[s][n][m] * y[s][m][e]     (WMMA, 1536 tile-waves)
__global__ void k_U(const _Float16* __restrict__ w_h, const _Float16* __restrict__ yT_h,
                    float* __restrict__ U) {
  int w = threadIdx.x >> 5, lane = threadIdx.x & 31;
  int gw = blockIdx.x * 8 + w;                  // 0..1535
  int s = gw / 192, r2 = gw % 192;
  int mt = r2 >> 2, nt = r2 & 3;
  int col = lane & 15, hi = lane >> 4;
  v8f acc = vzero8();
  const _Float16* A = w_h + (long)s * NMD;      // A(n, K=m), row-major stride MD
  const _Float16* Bt = yT_h + (long)s * DD * MD; // B(K=m, N=e) stored [e][m]
  for (int ks = 0; ks < 24; ++ks) {
    v16h a = load_a_frag(A, MD, mt * 16 + col, ks * 32, hi);
    v16h b = load_b_frag(Bt, MD, nt * 16 + col, ks * 32, hi);
    acc = wmma16(a, b, acc);
  }
#pragma unroll
  for (int r = 0; r < 8; ++r) {
    int n = mt * 16 + r + hi * 8;
    U[((long)s * ND + n) * DD + nt * 16 + col] = acc[r];
  }
}

// Solver: num = sx^T (W_s sy), model = num / (sum W_s + 1e-6)
__global__ void k_solver(const int* __restrict__ perm1, const int* __restrict__ perm2,
                         const _Float16* __restrict__ w_h,
                         const _Float16* __restrict__ sxT_g,
                         const _Float16* __restrict__ syT_g,
                         float* __restrict__ models, _Float16* __restrict__ model_hT) {
  __shared__ _Float16 Wc[32 * RP];   // mask chunk, A of step A
  __shared__ _Float16 Tt[DD * 32];   // T chunk stored [e][i_local], B of step B
  __shared__ float red[256];

  int blk = blockIdx.x;
  int s = blk / ITD, it = blk % ITD;
  int tid = threadIdx.x, w = tid >> 5, lane = tid & 31;
  int col = lane & 15, hi = lane >> 4;
  const int* p1 = perm1 + it * RN;
  const int* p2 = perm2 + it * RN;
  const _Float16* wbase = w_h + (long)s * NMD;
  const _Float16* sxT = sxT_g + (long)it * DD * RP;   // [d][i]
  const _Float16* syT = syT_g + (long)blk * DD * RP;  // [e][j]

  int mtA = w >> 2, ntA = w & 3;            // step A: 2x4 tiles of T (32 x 64)
  int t0 = 2 * w, t1 = 2 * w + 1;           // step B: 16 tiles of num (64 x 64)
  int tm0 = t0 >> 2, tn0 = t0 & 3, tm1 = t1 >> 2, tn1 = t1 & 3;

  v8f acc0 = vzero8(), acc1 = vzero8();
  float den = 0.0f;

  for (int chunk = 0; chunk < NCHUNK; ++chunk) {
    __syncthreads();
    // stage gathered mask chunk (rows i = chunk*32..+31, all j)
    for (int idx = tid; idx < 32 * RP; idx += 256) {
      int il = idx / RP, j = idx % RP;
      int i = chunk * 32 + il;
      float v = 0.0f;
      if (i < RN && j < RN) v = (float)wbase[(long)p1[i] * MD + p2[j]];
      Wc[idx] = (_Float16)v;
      den += v;
    }
    __syncthreads();
    // step A: T[i,e] = sum_j Wc[i,j] * sy[j,e]
    v8f at = vzero8();
    for (int ks = 0; ks < KSA; ++ks) {
      v16h a = load_a_frag(Wc, RP, mtA * 16 + col, ks * 32, hi);
      v16h b = load_b_frag(syT, RP, ntA * 16 + col, ks * 32, hi);
      at = wmma16(a, b, at);
    }
#pragma unroll
    for (int r = 0; r < 8; ++r) {
      int il = mtA * 16 + r + hi * 8;
      int e = ntA * 16 + col;
      Tt[e * 32 + il] = (_Float16)at[r];
    }
    __syncthreads();
    // step B: num[d,e] += sum_i sx[i,d] * T[i,e]   (A = sx^T)
    {
      v16h a = load_a_frag(sxT, RP, tm0 * 16 + col, chunk * 32, hi);
      v16h b = load_b_frag(Tt, 32, tn0 * 16 + col, 0, hi);
      acc0 = wmma16(a, b, acc0);
      v16h a2 = load_a_frag(sxT, RP, tm1 * 16 + col, chunk * 32, hi);
      v16h b2 = load_b_frag(Tt, 32, tn1 * 16 + col, 0, hi);
      acc1 = wmma16(a2, b2, acc1);
    }
  }
  __syncthreads();
  float dtot = block_reduce_256(den, red) + 1e-6f;

#pragma unroll
  for (int r = 0; r < 8; ++r) {
    int d0 = tm0 * 16 + r + hi * 8, e0 = tn0 * 16 + col;
    float v0 = acc0[r] / dtot;
    models[(long)blk * 4096 + d0 * 64 + e0] = v0;
    model_hT[(long)blk * 4096 + e0 * 64 + d0] = (_Float16)v0;
    int d1 = tm1 * 16 + r + hi * 8, e1 = tn1 * 16 + col;
    float v1 = acc1[r] / dtot;
    models[(long)blk * 4096 + d1 * 64 + e1] = v1;
    model_hT[(long)blk * 4096 + e1 * 64 + d1] = (_Float16)v1;
  }
}

// xt = x @ model; avg = (sum_n rc*||xt_n||^2 + bsum - 2*sum xt.U) / cnt
__global__ void k_avg(const _Float16* __restrict__ x_h, const _Float16* __restrict__ model_hT,
                      const float* __restrict__ U, const float* __restrict__ rc,
                      const float* __restrict__ bsum, const float* __restrict__ cnt,
                      _Float16* __restrict__ xt_h, float* __restrict__ avg) {
  __shared__ float red[256];
  int blk = blockIdx.x;
  int s = blk / ITD;
  int tid = threadIdx.x, w = tid >> 5, lane = tid & 31;
  int col = lane & 15, hi = lane >> 4;
  const _Float16* mh = model_hT + (long)blk * 4096;   // B(K=d', N=e) stored [e][d']
  float pa = 0.0f, pc = 0.0f;
  for (int tt = w; tt < 192; tt += 8) {               // 48 x 4 output tiles
    int mt = tt >> 2, nt = tt & 3;
    v8f acc = vzero8();
#pragma unroll
    for (int ks = 0; ks < 2; ++ks) {
      v16h a = load_a_frag(x_h, DD, mt * 16 + col, ks * 32, hi);
      v16h b = load_b_frag(mh, DD, nt * 16 + col, ks * 32, hi);
      acc = wmma16(a, b, acc);
    }
#pragma unroll
    for (int r = 0; r < 8; ++r) {
      int n = mt * 16 + r + hi * 8;
      int e = nt * 16 + col;
      float v = acc[r];
      xt_h[((long)blk * ND + n) * DD + e] = (_Float16)v;
      pa += rc[s * ND + n] * v * v;
      pc += v * U[((long)s * ND + n) * DD + e];
    }
  }
  float A = block_reduce_256(pa, red);
  float C = block_reduce_256(pc, red);
  if (tid == 0) avg[blk] = (A + bsum[s] - 2.0f * C) / cnt[s];
}

__global__ void k_best(const float* __restrict__ avg, int* __restrict__ best_it) {
  int s = threadIdx.x;
  if (s >= SD) return;
  int bi = 0;
  float bv = avg[s * ITD];
  for (int it = 1; it < ITD; ++it) {
    float v = avg[s * ITD + it];
    bool take = (v < bv) || ((v != v) && (bv == bv));  // NaN treated as minimum
    if (take) { bv = v; bi = it; }
  }
  best_it[s] = bi;
}

__global__ void k_xtn2(const _Float16* __restrict__ xt_h, const int* __restrict__ best_it,
                       float* __restrict__ xtn2) {
  int t = blockIdx.x * 256 + threadIdx.x;
  if (t >= SD * ND) return;
  int s = t / ND, n = t % ND;
  long blk = (long)s * ITD + best_it[s];
  float acc = 0.0f;
#pragma unroll 4
  for (int d = 0; d < DD; ++d) {
    float v = (float)xt_h[(blk * ND + n) * DD + d];
    acc += v * v;
  }
  xtn2[t] = acc;
}

// Full error grid for the winning model of each slice.
__global__ void k_err(const _Float16* __restrict__ xt_h, const _Float16* __restrict__ y_h,
                      const _Float16* __restrict__ w_h, const float* __restrict__ xtn2,
                      const float* __restrict__ ynorm, const int* __restrict__ best_it,
                      float* __restrict__ best_err) {
  int s = blockIdx.y;
  long blk = (long)s * ITD + best_it[s];
  int w = threadIdx.x >> 5, lane = threadIdx.x & 31;
  int tile = blockIdx.x * 8 + w;              // 0..2303 (48 x 48)
  int mt = tile / 48, tn = tile % 48;
  int col = lane & 15, hi = lane >> 4;
  const _Float16* A = xt_h + blk * ND * DD;   // A(n, K=e)
  const _Float16* Bt = y_h + (long)s * MD * DD; // B(K=e, N=m) stored [m][e]
  v8f acc = vzero8();
#pragma unroll
  for (int ks = 0; ks < 2; ++ks) {
    v16h a = load_a_frag(A, DD, mt * 16 + col, ks * 32, hi);
    v16h b = load_b_frag(Bt, DD, tn * 16 + col, ks * 32, hi);
    acc = wmma16(a, b, acc);
  }
#pragma unroll
  for (int r = 0; r < 8; ++r) {
    int n = mt * 16 + r + hi * 8;
    int m = tn * 16 + col;
    float e = xtn2[s * ND + n] + ynorm[s * MD + m] - 2.0f * acc[r];
    float wv = (float)w_h[((long)s * ND + n) * MD + m];
    best_err[((long)s * ND + n) * MD + m] = (wv != 0.0f) ? e : nanf_dev();
  }
}

// Exact nanquantile(0.75, linear) via two 4-pass radix selects on sortable bits.
__global__ void k_quant(const float* __restrict__ best_err, const float* __restrict__ cnt,
                        float* __restrict__ thr) {
  __shared__ unsigned s_hist[256];
  __shared__ unsigned s_prefix;
  __shared__ int s_rank;
  int s = blockIdx.x, tid = threadIdx.x;
  const float* data = best_err + (long)s * NMD;
  int cntv = (int)(cnt[s] + 0.5f);
  if (cntv < 1) { if (tid == 0) thr[s] = nanf_dev(); return; }
  int num = 3 * (cntv - 1);
  int k0 = num >> 2;
  int rem = num & 3;
  int k1 = k0 + (rem ? 1 : 0);
  float frac = (float)rem * 0.25f;

  float vsel[2];
  int ranks[2] = {k0, k1};
  for (int ri = 0; ri < 2; ++ri) {
    if (tid == 0) { s_prefix = 0u; s_rank = ranks[ri]; }
    __syncthreads();
    for (int pass = 3; pass >= 0; --pass) {
      s_hist[tid] = 0u;
      __syncthreads();
      unsigned pref = s_prefix;
      for (int idx = tid; idx < NMD; idx += 256) {
        unsigned b = __float_as_uint(data[idx]);
        unsigned key = b ^ (((int)b < 0) ? 0xFFFFFFFFu : 0x80000000u);
        bool match = (pass == 3) || ((key >> ((pass + 1) * 8)) == pref);
        if (match) atomicAdd(&s_hist[(key >> (pass * 8)) & 255u], 1u);
      }
      __syncthreads();
      if (tid == 0) {
        unsigned cum = 0, bin = 255;
        int rk = s_rank;
        for (unsigned bp = 0; bp < 256; ++bp) {
          unsigned c = s_hist[bp];
          if (cum + c > (unsigned)rk) { bin = bp; break; }
          cum += c;
        }
        s_rank = rk - (int)cum;
        s_prefix = (s_prefix << 8) | bin;
      }
      __syncthreads();
    }
    unsigned key = s_prefix;
    unsigned b = key ^ ((key & 0x80000000u) ? 0x80000000u : 0xFFFFFFFFu);
    vsel[ri] = __uint_as_float(b);
    __syncthreads();
  }
  if (tid == 0) thr[s] = vsel[0] + frac * (vsel[1] - vsel[0]);
}

__global__ void k_order(const float* __restrict__ avg, const int* __restrict__ best_it,
                        int* __restrict__ order, float* __restrict__ out_topk) {
  if (threadIdx.x != 0 || blockIdx.x != 0) return;
  float av[SD];
  bool used[SD];
  for (int s = 0; s < SD; ++s) { av[s] = avg[s * ITD + best_it[s]]; used[s] = false; }
  for (int k = 0; k < SD; ++k) {
    int bi = -1;
    float bv = 0.0f;
    for (int s = 0; s < SD; ++s) {
      if (used[s]) continue;
      if (bi < 0 || av[s] > bv) { bi = s; bv = av[s]; }  // strict > keeps lowest idx on ties
    }
    used[bi] = true;
    order[k] = bi;
    out_topk[k] = bv;
  }
}

__global__ void k_tm(const float* __restrict__ models, const int* __restrict__ best_it,
                     const int* __restrict__ order, float* __restrict__ out_tm) {
  int k = blockIdx.x;
  int s = order[k];
  long blk = (long)s * ITD + best_it[s];
  for (int idx = threadIdx.x; idx < DD * DD; idx += 256)
    out_tm[(long)k * DD * DD + idx] = models[blk * DD * DD + idx];
}

__global__ void k_ti(const float* __restrict__ best_err, const float* __restrict__ thr,
                     const int* __restrict__ order, float* __restrict__ out_ti) {
  long t = (long)blockIdx.x * 256 + threadIdx.x;
  if (t >= (long)SD * NMD) return;
  int k = (int)(t / NMD);
  long r = t % NMD;
  int s = order[k];
  float v = best_err[(long)s * NMD + r];
  out_ti[t] = (v < thr[s]) ? 1.0f : 0.0f;   // NaN < thr -> false, matches reference
}

// ---------------------------------------------------------------------------
extern "C" void kernel_launch(void* const* d_in, const int* in_sizes, int n_in,
                              void* d_out, int out_size, void* d_ws, size_t ws_size,
                              hipStream_t stream) {
  const float* qk = (const float*)d_in[0];       // (1,768,64)
  const float* kks = (const float*)d_in[1];      // (1,8,768,64)
  const int* qm = (const int*)d_in[2];           // (8,768,768) bool -> int
  const int* km = (const int*)d_in[3];
  const int* perm1 = (const int*)d_in[4];        // (16,460)
  const int* perm2 = (const int*)d_in[5];

  char* p = (char*)d_ws;
  size_t off = 0;
  auto grab = [&](size_t bytes) {
    void* r = p + off;
    off = (off + bytes + 255) & ~(size_t)255;
    return r;
  };
  _Float16* x_h   = (_Float16*)grab(sizeof(_Float16) * ND * DD);
  _Float16* y_h   = (_Float16*)grab(sizeof(_Float16) * SD * MD * DD);
  _Float16* yT_h  = (_Float16*)grab(sizeof(_Float16) * SD * DD * MD);
  float*    ynorm = (float*)grab(sizeof(float) * SD * MD);
  _Float16* w_h   = (_Float16*)grab(sizeof(_Float16) * (size_t)SD * NMD);
  float*    rc    = (float*)grab(sizeof(float) * SD * ND);
  float*    cc    = (float*)grab(sizeof(float) * SD * MD);
  float*    cnt   = (float*)grab(sizeof(float) * SD);
  float*    bsum  = (float*)grab(sizeof(float) * SD);
  float*    U     = (float*)grab(sizeof(float) * SD * ND * DD);
  _Float16* sxT_g = (_Float16*)grab(sizeof(_Float16) * ITD * DD * RP);
  _Float16* syT_g = (_Float16*)grab(sizeof(_Float16) * (size_t)SD * ITD * DD * RP);
  float*    models  = (float*)grab(sizeof(float) * SD * ITD * DD * DD);
  _Float16* model_hT = (_Float16*)grab(sizeof(_Float16) * SD * ITD * DD * DD);
  _Float16* xt_h  = (_Float16*)grab(sizeof(_Float16) * (size_t)SD * ITD * ND * DD);
  float*    avg   = (float*)grab(sizeof(float) * SD * ITD);
  int*      best_it = (int*)grab(sizeof(int) * SD);
  float*    xtn2  = (float*)grab(sizeof(float) * SD * ND);
  float*    best_err = (float*)grab(sizeof(float) * (size_t)SD * NMD);
  float*    thr   = (float*)grab(sizeof(float) * SD);
  int*      order = (int*)grab(sizeof(int) * SD);

  float* out = (float*)d_out;
  const long TI_SZ = (long)SD * NMD;        // 4,718,592
  const long TM_SZ = (long)SD * DD * DD;    // 32,768
  float* out_ti = out;
  float* out_tm = out + TI_SZ;
  float* out_err = out + TI_SZ + TM_SZ;

  k_pre1<<<(SD * MD + ND + 255) / 256, 256, 0, stream>>>(qk, kks, x_h, y_h, yT_h, ynorm);
  k_mask<<<(SD * ND + 255) / 256, 256, 0, stream>>>(qm, km, w_h, rc);
  k_cols<<<(SD * MD + 255) / 256, 256, 0, stream>>>(w_h, cc);
  k_sums<<<SD, 256, 0, stream>>>(rc, cc, ynorm, cnt, bsum);
  {
    int total = ITD * DD * RP + SD * ITD * DD * RP;
    k_gather<<<(total + 255) / 256, 256, 0, stream>>>(perm1, perm2, x_h, y_h, sxT_g, syT_g);
  }
  k_U<<<192, 256, 0, stream>>>(w_h, yT_h, U);
  k_solver<<<SD * ITD, 256, 0, stream>>>(perm1, perm2, w_h, sxT_g, syT_g, models, model_hT);
  k_avg<<<SD * ITD, 256, 0, stream>>>(x_h, model_hT, U, rc, bsum, cnt, xt_h, avg);
  k_best<<<1, 32, 0, stream>>>(avg, best_it);
  k_xtn2<<<(SD * ND + 255) / 256, 256, 0, stream>>>(xt_h, best_it, xtn2);
  k_err<<<dim3(288, SD), 256, 0, stream>>>(xt_h, y_h, w_h, xtn2, ynorm, best_it, best_err);
  k_quant<<<SD, 256, 0, stream>>>(best_err, cnt, thr);
  k_order<<<1, 32, 0, stream>>>(avg, best_it, order, out_err);
  k_tm<<<SD, 256, 0, stream>>>(models, best_it, order, out_tm);
  k_ti<<<(int)((TI_SZ + 255) / 256), 256, 0, stream>>>(best_err, thr, order, out_ti);
}
